// RNN_18021682774382
// MI455X (gfx1250) — matrix operand order repaired and checked by reading, not compiled, not run on hardware
//
#include <hip/hip_runtime.h>
#include <hip/hip_bf16.h>
#include <math.h>

// ---------------------------------------------------------------------------
// RNN forward for MI455X (gfx1250, wave32, WMMA).
//   B=64, T=512, E=768, H=256, V=3072
//   out  = [64,3072] f32, hidden = [64,256] f32  (concatenated in d_out)
// Strategy:
//   k0: convert Wxh, Wy to bf16 (L2-resident reuse)
//   k1: xw[t,b,:] = bf16(emb[idx[b,t]]) @ bf16(Wxh)   (WMMA, LDS-staged A)
//   k2: 512 serial steps, h = tanh(x_t + h@Whh + Bh)  (16 waves x 16 cols,
//       Whh cols persistent in VGPRs, h double-buffered in LDS, v_tanh_f32,
//       1 barrier/step)
//   k3: out = h @ Wy + By                             (WMMA)
// ---------------------------------------------------------------------------

typedef __attribute__((ext_vector_type(16))) __bf16 v16bf;
typedef __attribute__((ext_vector_type(8)))  float  v8f;

#if __has_builtin(__builtin_amdgcn_tanhf)
#define FAST_TANH(x) __builtin_amdgcn_tanhf(x)   // v_tanh_f32 (gfx1250 trans op)
#else
#define FAST_TANH(x) tanhf(x)
#endif

// A fragment, 16x32 bf16 (M x K). Lane l: row m = l&15.
// elems 0..7  : K = kb + (l>=16 ? 8 : 0) + i
// elems 8..15 : K = kb + 16 + (l>=16 ? 8 : 0) + (i-8)
__device__ __forceinline__ v16bf ld_frag_a(const __bf16* base, int ld, int lane, int kb) {
  const __bf16* p = base + (size_t)(lane & 15) * ld + kb + ((lane >> 4) << 3);
  v16bf r;
  ((uint4*)&r)[0] = *(const uint4*)(p);
  ((uint4*)&r)[1] = *(const uint4*)(p + 16);
  return r;
}

// B fragment, 32x16 bf16 (K x N). Lane l: col N = nc + (l&15); K same order as A fragment.
__device__ __forceinline__ v16bf ld_frag_b(const __bf16* base, int ld, int lane, int kb, int nc) {
  const __bf16* p = base + (size_t)(kb + ((lane >> 4) << 3)) * ld + nc + (lane & 15);
  v16bf r;
  #pragma unroll
  for (int i = 0; i < 8; ++i) {
    r[i]     = p[(size_t)i * ld];
    r[i + 8] = p[(size_t)(16 + i) * ld];
  }
  return r;
}

__device__ __forceinline__ v8f wmma_bf16(v16bf a, v16bf b, v8f c) {
  // (neg_a, A, neg_b, B, c_mod, C, reuse_a, reuse_b)
  return __builtin_amdgcn_wmma_f32_16x16x32_bf16(false, a, false, b, (short)0, c,
                                                 false, false);
}

// --------------------------- k0: f32 -> bf16 -------------------------------
__global__ void __launch_bounds__(256) cvt_bf16_kernel(const float* __restrict__ src,
                                                       __bf16* __restrict__ dst, int n) {
  int i = blockIdx.x * 256 + threadIdx.x;
  if (i < n) dst[i] = (__bf16)src[i];
}

// ------------------- k1: xw = gather(emb) @ Wxh ----------------------------
// grid.x = 2048 (M-tiles of 16 over M = T*B = 32768, m = t*64 + b)
// block  = 256 (8 waves); wave w covers cols [w*32, w*32+32)
__global__ void __launch_bounds__(256) xwxh_gemm_kernel(
    const int*    __restrict__ idx,   // [64,512]
    const float*  __restrict__ emb,   // [3072,768]
    const __bf16* __restrict__ WxhB,  // [768,256] bf16
    float*        __restrict__ xw)    // [32768,256] f32
{
  __shared__ __bf16 sA[16 * 776];     // 16 rows x 768 (pad->776 vs bank conflicts)
  const int tid  = threadIdx.x;
  const int lane = tid & 31;
  const int wave = tid >> 5;
  const int mr   = blockIdx.x * 16;

  // Stage gathered embedding rows into LDS as bf16 (2 rows per wave).
  #pragma unroll
  for (int rr = 0; rr < 2; ++rr) {
    const int row = wave * 2 + rr;
    const int m = mr + row;
    const int t = m >> 6, b = m & 63;
    const float* src = emb + (size_t)idx[b * 512 + t] * 768;
    #pragma unroll
    for (int e = lane * 4; e < 768; e += 128) {
      float4 v = *(const float4*)(src + e);
      __bf16 h4[4] = {(__bf16)v.x, (__bf16)v.y, (__bf16)v.z, (__bf16)v.w};
      *(uint2*)&sA[row * 776 + e] = *(const uint2*)h4;
    }
  }
  __syncthreads();

  const int nc0 = wave * 32;
  v8f acc0 = {}, acc1 = {};
  #pragma unroll 4
  for (int kb = 0; kb < 768; kb += 32) {
    v16bf a  = ld_frag_a(sA, 776, lane, kb);
    v16bf b0 = ld_frag_b(WxhB, 256, lane, kb, nc0);
    v16bf b1 = ld_frag_b(WxhB, 256, lane, kb, nc0 + 16);
    acc0 = wmma_bf16(a, b0, acc0);
    acc1 = wmma_bf16(a, b1, acc1);
  }

  // C/D layout: m = r + 8*(lane>=16), n = lane&15
  const int mb = (lane >> 4) << 3;
  const int n0 = nc0 + (lane & 15);
  float* op = xw + (size_t)mr * 256;
  #pragma unroll
  for (int r = 0; r < 8; ++r) {
    op[(size_t)(mb + r) * 256 + n0]      = acc0[r];
    op[(size_t)(mb + r) * 256 + n0 + 16] = acc1[r];
  }
}

// -------------------- k2: serial recurrence over T -------------------------
// grid.x = 4 (batch tiles of 16), block = 512 (16 waves, 16 cols each).
// Per wave per step: 8 x-loads (overlap WMMA chain), 16 ds_load_b128,
// 8 WMMAs (serial on acc), 8 v_tanh, 8 ds_store_b16, 1 barrier.
__global__ void __launch_bounds__(512) rnn_recurrence_kernel(
    const float* __restrict__ xw,   // [512,64,256] f32
    const float* __restrict__ Whh,  // [256,256] f32
    const float* __restrict__ Bh,   // [256] f32
    float*       __restrict__ hid,  // [64,256] f32 (final hidden, d_out)
    __bf16*      __restrict__ hF)   // [64,256] bf16 (for output GEMM)
{
  __shared__ __bf16 sH[2][16 * 264];   // double-buffered h tile (pad 256->264)
  const int tid   = threadIdx.x;
  const int lane  = tid & 31;
  const int wave  = tid >> 5;          // 0..15
  const int bbase = blockIdx.x * 16;
  const int nc    = wave * 16;         // this wave's 16-column N-tile

  // Persistent Whh B-fragments in VGPRs: 8 k-steps x 8 VGPRs = 64 VGPRs.
  v16bf bw[8];
  #pragma unroll
  for (int kk = 0; kk < 8; ++kk) {
    const float* p = Whh + (size_t)(kk * 32 + ((lane >> 4) << 3)) * 256 + nc + (lane & 15);
    v16bf b;
    #pragma unroll
    for (int i = 0; i < 16; ++i) b[i] = (__bf16)p[(size_t)((i & 7) + ((i >> 3) << 4)) * 256];
    bw[kk] = b;
  }

  const int mb = (lane >> 4) << 3;        // C-frag row base
  const int nl = lane & 15;               // C-frag col within tile
  const float bh = Bh[nc + nl];

  // h0 = 0
  for (int i = tid; i < 16 * 264; i += 512) sH[0][i] = (__bf16)0.0f;
  __syncthreads();

  int cur = 0;
  for (int t = 0; t < 512; ++t) {
    // Issue x_t loads early; consumed only after the WMMA chain.
    const float* xp = xw + (size_t)(t * 64 + bbase) * 256 + nc + nl;
    float xv[8];
    #pragma unroll
    for (int r = 0; r < 8; ++r) xv[r] = xp[(size_t)(mb + r) * 256];

    // h @ Whh  (A from LDS, B persistent in VGPRs)
    v8f acc = {};
    #pragma unroll
    for (int kk = 0; kk < 8; ++kk) {
      v16bf a = ld_frag_a(&sH[cur][0], 264, lane, kk * 32);
      acc = wmma_bf16(a, bw[kk], acc);
    }

    // h' = tanh(h@Whh + x_t + Bh) -> LDS next buffer (+ final outputs)
    const int nxt = cur ^ 1;
    #pragma unroll
    for (int r = 0; r < 8; ++r) {
      float v = FAST_TANH(acc[r] + xv[r] + bh);
      sH[nxt][(mb + r) * 264 + nc + nl] = (__bf16)v;
      if (t == 511) {
        hid[(size_t)(bbase + mb + r) * 256 + nc + nl] = v;
        hF [(size_t)(bbase + mb + r) * 256 + nc + nl] = (__bf16)v;
      }
    }
    __syncthreads();
    cur = nxt;
  }
}

// ------------------------- k3: out = h @ Wy + By ---------------------------
// grid = (4, 12): 4 M-tiles x 12 N-chunks(256); block = 256 (8 waves x 32 cols)
__global__ void __launch_bounds__(256) out_gemm_kernel(
    const __bf16* __restrict__ hF,   // [64,256] bf16
    const __bf16* __restrict__ WyB,  // [256,3072] bf16
    const float*  __restrict__ By,   // [3072]
    float*        __restrict__ out)  // [64,3072]
{
  const int tid  = threadIdx.x;
  const int lane = tid & 31;
  const int wave = tid >> 5;
  const int mr   = blockIdx.x * 16;
  const int nc0  = blockIdx.y * 256 + wave * 32;

  v8f acc0 = {}, acc1 = {};
  #pragma unroll
  for (int kb = 0; kb < 256; kb += 32) {
    v16bf a  = ld_frag_a(hF + (size_t)mr * 256, 256, lane, kb);
    v16bf b0 = ld_frag_b(WyB, 3072, lane, kb, nc0);
    v16bf b1 = ld_frag_b(WyB, 3072, lane, kb, nc0 + 16);
    acc0 = wmma_bf16(a, b0, acc0);
    acc1 = wmma_bf16(a, b1, acc1);
  }

  const int mb = (lane >> 4) << 3;
  const int nl = lane & 15;
  const float by0 = By[nc0 + nl];
  const float by1 = By[nc0 + 16 + nl];
  #pragma unroll
  for (int r = 0; r < 8; ++r) {
    out[(size_t)(mr + mb + r) * 3072 + nc0 + nl]      = acc0[r] + by0;
    out[(size_t)(mr + mb + r) * 3072 + nc0 + 16 + nl] = acc1[r] + by1;
  }
}

// ---------------------------------------------------------------------------
extern "C" void kernel_launch(void* const* d_in, const int* in_sizes, int n_in,
                              void* d_out, int out_size, void* d_ws, size_t ws_size,
                              hipStream_t stream) {
  const int*   idx = (const int*)d_in[0];    // [64,512]
  const float* emb = (const float*)d_in[1];  // [3072,768]
  const float* Wxh = (const float*)d_in[2];  // [768,256]
  const float* Whh = (const float*)d_in[3];  // [256,256]
  const float* Wy  = (const float*)d_in[4];  // [256,3072]
  const float* By  = (const float*)d_in[5];  // [3072]
  const float* Bh  = (const float*)d_in[6];  // [256]

  float* out = (float*)d_out;                // [64,3072]
  float* hid = out + 64 * 3072;              // [64,256]

  char* ws = (char*)d_ws;
  __bf16* WxhB = (__bf16*)(ws);                                    // 768*256
  __bf16* WyB  = (__bf16*)(ws + 393216);                           // 256*3072
  __bf16* hF   = (__bf16*)(ws + 393216 + 1572864);                 // 64*256
  float*  xw   = (float*)(ws + 393216 + 1572864 + 32768);          // 32768*256 f32

  cvt_bf16_kernel<<<768,  256, 0, stream>>>(Wxh, WxhB, 768 * 256);
  cvt_bf16_kernel<<<3072, 256, 0, stream>>>(Wy,  WyB,  256 * 3072);
  xwxh_gemm_kernel<<<2048, 256, 0, stream>>>(idx, emb, WxhB, xw);
  rnn_recurrence_kernel<<<4, 512, 0, stream>>>(xw, Whh, Bh, hid, hF);
  out_gemm_kernel<<<dim3(4, 12), 256, 0, stream>>>(hF, WyB, By, out);
}
